// GCN_27797028339956
// MI455X (gfx1250) — matrix operand order repaired and checked by reading, not compiled, not run on hardware
//
#include <hip/hip_runtime.h>

// GCN 2-layer forward for MI455X (gfx1250, wave32).
// Dense transforms: v_wmma_f32_16x16x32_bf16 with pre-converted bf16 operands
// (A: bf16 X/H rows, B: bf16 W pre-transposed) -> hot loop is b128 loads + wmma.
// Edge aggregation: hardware global_atomic_add_f32 into L2-resident AGG.

typedef __attribute__((ext_vector_type(16))) __bf16 v16bf;
typedef __attribute__((ext_vector_type(8)))  __bf16 v8bf;
typedef __attribute__((ext_vector_type(8)))  float  v8f;

// ---------------------------------------------------------------------------
// Degree / normalization
// ---------------------------------------------------------------------------
__global__ void deg_init(float* __restrict__ deg, int n) {
  int i = blockIdx.x * blockDim.x + threadIdx.x;
  if (i < n) deg[i] = 1.0f;            // self-loop contribution
}

__global__ void deg_count(const int* __restrict__ coli, float* __restrict__ deg, int ne) {
  int i = blockIdx.x * blockDim.x + threadIdx.x;
  if (i < ne) unsafeAtomicAdd(&deg[coli[i]], 1.0f);
}

__global__ void deg_to_dinv(float* __restrict__ deg, int n) {
  int i = blockIdx.x * blockDim.x + threadIdx.x;
  if (i < n) deg[i] = rsqrtf(deg[i]);  // deg >= 1 always (self loops)
}

// ---------------------------------------------------------------------------
// One-shot operand prep
// ---------------------------------------------------------------------------
// f32 -> bf16, 8 elements per thread (float4 x2 in, one b128 out).
__global__ void f32_to_bf16_vec(const float* __restrict__ src, __bf16* __restrict__ dst,
                                int n8) {
  int i = blockIdx.x * blockDim.x + threadIdx.x;
  if (i < n8) {
    const float4 a = ((const float4*)src)[2 * i];
    const float4 b = ((const float4*)src)[2 * i + 1];
    v8bf o;
    o[0] = (__bf16)a.x; o[1] = (__bf16)a.y; o[2] = (__bf16)a.z; o[3] = (__bf16)a.w;
    o[4] = (__bf16)b.x; o[5] = (__bf16)b.y; o[6] = (__bf16)b.z; o[7] = (__bf16)b.w;
    ((v8bf*)dst)[i] = o;
  }
}

// W [K, Ncols] f32 row-major  ->  Wt [Ncols, K] bf16 row-major (transpose+convert)
__global__ void w_transpose_bf16(const float* __restrict__ W, __bf16* __restrict__ Wt,
                                 int K, int Ncols) {
  int i = blockIdx.x * blockDim.x + threadIdx.x;   // i = n*K + k
  if (i < K * Ncols) {
    int n = i / K, k = i - n * K;
    Wt[i] = (__bf16)W[(size_t)k * Ncols + n];
  }
}

// ---------------------------------------------------------------------------
// WMMA GEMM: H = X @ W (bf16 in, f32 out) and AGG = H * dinv^2 (self-loop init)
// One wave -> one 16x16 tile; block covers 16 rows x NDIM cols; the shared
// 16xKDIM A tile (4 KB bf16) stays hot in the WGP$ across the block's waves.
// rows % 16 == 0 (100000 = 6250*16) so EXEC is all-ones at WMMA.
// ---------------------------------------------------------------------------
template <int KDIM, int NDIM>
__global__ __launch_bounds__(32 * (NDIM / 16))
void gcn_gemm_wmma(const __bf16* __restrict__ X, const __bf16* __restrict__ Wt,
                   const float* __restrict__ dinv,
                   float* __restrict__ H, float* __restrict__ AGG)
{
  const int wave  = threadIdx.x >> 5;
  const int lane  = threadIdx.x & 31;
  const int r     = lane & 15;
  const int half  = lane >> 4;
  const int mbase = blockIdx.x * 16;
  const int nbase = wave * 16;

  // A (16-bit 16x32, ISA 7.12.2): lane-half selects K-octets {0..7,16..23} vs
  // {8..15,24..31}; both octets are 16 B contiguous in bf16 memory.
  const __bf16* xr = X  + (size_t)(mbase + r) * KDIM + half * 8;
  // B (32x16): lane = column n, lane-half selects K-octets {0..7,16..23} vs
  // {8..15,24..31} (same as A); with Wt pre-transposed each octet is 16 B contiguous.
  const __bf16* wr = Wt + (size_t)(nbase + r) * KDIM + half * 8;

  v8f c = {};
#pragma unroll
  for (int k0 = 0; k0 < KDIM; k0 += 32) {
    v8bf alo = *(const v8bf*)(xr + k0);
    v8bf ahi = *(const v8bf*)(xr + k0 + 16);
    v8bf blo = *(const v8bf*)(wr + k0);
    v8bf bhi = *(const v8bf*)(wr + k0 + 16);
    v16bf a = __builtin_shufflevector(alo, ahi, 0,1,2,3,4,5,6,7,8,9,10,11,12,13,14,15);
    v16bf b = __builtin_shufflevector(blo, bhi, 0,1,2,3,4,5,6,7,8,9,10,11,12,13,14,15);
    c = __builtin_amdgcn_wmma_f32_16x16x32_bf16(false, a, false, b,
                                                (short)0, c, false, false);
  }

  // C/D layout: VGPR i holds M = 8*half + i, N = lane&15.
#pragma unroll
  for (int i = 0; i < 8; ++i) {
    const int   m  = mbase + 8 * half + i;
    const int   n  = nbase + r;
    const float v  = c[i];
    const float di = dinv[m];
    H  [(size_t)m * NDIM + n] = v;
    AGG[(size_t)m * NDIM + n] = v * di * di;   // self-loop term
  }
}

// ---------------------------------------------------------------------------
// Edge scatter: AGG[col] += H[row] * dinv[row]*dinv[col]
// F/4 lanes per edge, float4 gathers, hardware f32 atomics into L2.
// ---------------------------------------------------------------------------
template <int F>
__global__ __launch_bounds__(256)
void edge_scatter(const int* __restrict__ rowi, const int* __restrict__ coli,
                  const float* __restrict__ dinv,
                  const float* __restrict__ H, float* __restrict__ AGG, int ne)
{
  constexpr int LPE = F / 4;                 // lanes per edge
  const int slot   = threadIdx.x / LPE;
  const int fl     = threadIdx.x % LPE;
  const int spb    = blockDim.x / LPE;       // edges per block
  const int stride = gridDim.x * spb;

  for (int e = blockIdx.x * spb + slot; e < ne; e += stride) {
    const int   rs  = rowi[e];
    const int   cd  = coli[e];
    const float nrm = dinv[rs] * dinv[cd];
    const float4 hv = *(const float4*)(H + (size_t)rs * F + fl * 4);

    // Warm caches for the next edge's source row (global_prefetch_b8).
    const int en = e + stride;
    if (en < ne)
      __builtin_prefetch(H + (size_t)rowi[en] * F + fl * 4, 0, 1);

    float* dst = AGG + (size_t)cd * F + fl * 4;
    unsafeAtomicAdd(dst + 0, hv.x * nrm);
    unsafeAtomicAdd(dst + 1, hv.y * nrm);
    unsafeAtomicAdd(dst + 2, hv.z * nrm);
    unsafeAtomicAdd(dst + 3, hv.w * nrm);
  }
}

// ---------------------------------------------------------------------------
// Layer-1 epilogue: bf16(out) = relu(AGG + b1)   (feeds GEMM2's A operand)
// ---------------------------------------------------------------------------
template <int F>
__global__ void bias_relu_to_bf16(const float* __restrict__ A, const float* __restrict__ b,
                                  __bf16* __restrict__ O, int total) {
  int i = blockIdx.x * blockDim.x + threadIdx.x;
  if (i < total) {
    float v = A[i] + b[i & (F - 1)];
    O[i] = (__bf16)fmaxf(v, 0.0f);
  }
}

// Final epilogue: out += b2 (in place, f32)
template <int F>
__global__ void bias_add(float* __restrict__ A, const float* __restrict__ b, int total) {
  int i = blockIdx.x * blockDim.x + threadIdx.x;
  if (i < total) A[i] = A[i] + b[i & (F - 1)];
}

// ---------------------------------------------------------------------------
extern "C" void kernel_launch(void* const* d_in, const int* in_sizes, int n_in,
                              void* d_out, int out_size, void* d_ws, size_t ws_size,
                              hipStream_t stream)
{
  (void)n_in; (void)out_size; (void)ws_size;

  const float* x   = (const float*)d_in[0];
  const int*   ei  = (const int*)  d_in[1];
  const float* W1  = (const float*)d_in[2];
  const float* b1  = (const float*)d_in[3];
  const float* W2  = (const float*)d_in[4];
  const float* b2  = (const float*)d_in[5];
  float*       out = (float*)d_out;

  constexpr int IN = 128, HID = 128, OUTF = 64;
  const int nN = in_sizes[0] / IN;   // 100000 (multiple of 16)
  const int nE = in_sizes[1] / 2;    // 1600000
  const int* rowi = ei;              // edge_index[0] = sources
  const int* coli = ei + nE;         // edge_index[1] = targets

  // Workspace: dinv | H (f32) | AGG1 (f32) | Xbf (bf16) | W1t | W2t  (~129 MB)
  char* ws = (char*)d_ws;
  auto alignup = [](size_t v) { return (v + 255) & ~(size_t)255; };
  size_t off = 0;
  float*  dinv = (float*) (ws + off); off += alignup((size_t)nN * 4);
  float*  H    = (float*) (ws + off); off += alignup((size_t)nN * HID * 4);
  float*  AGG1 = (float*) (ws + off); off += alignup((size_t)nN * HID * 4);
  __bf16* Xbf  = (__bf16*)(ws + off); off += alignup((size_t)nN * IN * 2);
  __bf16* W1t  = (__bf16*)(ws + off); off += alignup((size_t)IN * HID * 2);
  __bf16* W2t  = (__bf16*)(ws + off);
  float*  H2   = H;                  // H dead after layer-1 scatter
  __bf16* Hbf  = Xbf;                // Xbf dead after GEMM1

  const dim3 b256(256);

  // --- normalization -------------------------------------------------------
  deg_init   <<<(nN + 255) / 256, b256, 0, stream>>>(dinv, nN);
  deg_count  <<<(nE + 255) / 256, b256, 0, stream>>>(coli, dinv, nE);
  deg_to_dinv<<<(nN + 255) / 256, b256, 0, stream>>>(dinv, nN);

  // --- operand prep --------------------------------------------------------
  {
    int n8 = nN * IN / 8;
    f32_to_bf16_vec<<<(n8 + 255) / 256, b256, 0, stream>>>(x, Xbf, n8);
    w_transpose_bf16<<<(IN * HID + 255) / 256, b256, 0, stream>>>(W1, W1t, IN, HID);
    w_transpose_bf16<<<(HID * OUTF + 255) / 256, b256, 0, stream>>>(W2, W2t, HID, OUTF);
  }

  // --- layer 1: h = relu(Ahat (x@W1) + b1) ---------------------------------
  gcn_gemm_wmma<IN, HID><<<nN / 16, dim3(256), 0, stream>>>(Xbf, W1t, dinv, H, AGG1);
  {
    constexpr int spb = 256 / (HID / 4);          // 8 edges / block
    edge_scatter<HID><<<(nE + spb - 1) / spb, b256, 0, stream>>>(
        rowi, coli, dinv, H, AGG1, nE);
  }
  bias_relu_to_bf16<HID><<<((size_t)nN * HID + 255) / 256, b256, 0, stream>>>(
      AGG1, b1, Hbf, nN * HID);

  // --- layer 2: out = Ahat (h@W2) + b2  (d_out doubles as AGG2) ------------
  gcn_gemm_wmma<HID, OUTF><<<nN / 16, dim3(128), 0, stream>>>(Hbf, W2t, dinv, H2, out);
  {
    constexpr int spb = 256 / (OUTF / 4);         // 16 edges / block
    edge_scatter<OUTF><<<(nE + spb - 1) / spb, b256, 0, stream>>>(
        rowi, coli, dinv, H2, out, nE);
  }
  bias_add<OUTF><<<((size_t)nN * OUTF + 255) / 256, b256, 0, stream>>>(
      out, b2, nN * OUTF);
}
